// CausalMemory_3255585210490
// MI455X (gfx1250) — matrix operand order repaired and checked by reading, not compiled, not run on hardware
//
#include <hip/hip_runtime.h>
#include <hip/hip_bf16.h>

// ---------------------------------------------------------------------------
// CausalMemory on MI455X (gfx1250, wave32, WMMA bf16 16x16x32, f32 accumulate)
// Round 2: global->LDS staging now uses CDNA5 async-DMA
// (global_load_async_to_lds_b128, ASYNCcnt) with double-buffered LDS tiles so
// the copies overlap WMMA compute.
//
// Workspace layout (requires ws_size >= ~82 MB):
//   wqT,wkT,wvT,woT : 4 x 512x512 bf16 (pre-transposed: Wt[n][k] = W[k][n])
//   x_bf            : [B*T][C] bf16
//   q_bf, k_bf      : [B*T][C] bf16
//   v_t             : [B][C][T] bf16  (transposed for B-operand loads)
//   r_bf            : [B*T][C] bf16  (retrieved)
// ---------------------------------------------------------------------------

#define T_SEQ 4096
#define CDIM  512
#define BATCH 4

typedef __attribute__((ext_vector_type(16))) __bf16       bf16x16;
typedef __attribute__((ext_vector_type(8)))  float        f32x8;
typedef __attribute__((ext_vector_type(4)))  float        f32x4;
typedef __attribute__((ext_vector_type(4)))  unsigned int u32x4;

union FragU { bf16x16 v; u32x4 q[2]; };

__device__ __forceinline__ unsigned short f2bf(float f) {
  unsigned int u = __builtin_bit_cast(unsigned int, f);
  u += 0x7FFFu + ((u >> 16) & 1u);   // round-to-nearest-even
  return (unsigned short)(u >> 16);
}

__device__ __forceinline__ f32x8 wmma_bf16(bf16x16 a, bf16x16 b, f32x8 c) {
  return __builtin_amdgcn_wmma_f32_16x16x32_bf16(false, a, false, b,
                                                 (short)0, c, false, false);
}

// 16-byte per-lane async DMA: global memory -> LDS, tracked by ASYNCcnt.
// LDS byte address = low 32 bits of the generic pointer (ISA: LDS aperture
// truncates to addr[31:0]).
__device__ __forceinline__ void async_copy16(void* lds_dst, const void* gsrc) {
  unsigned loff = (unsigned)(unsigned long long)lds_dst;
  asm volatile("global_load_async_to_lds_b128 %0, %1, off"
               :: "v"(loff), "v"(gsrc) : "memory");
}
#define WAIT_ASYNC(imm) asm volatile("s_wait_asynccnt " imm ::: "memory")

// A-operand (16x32 bf16) from a row-major [M][K] tile.  lane: m=lane&15,
// h=lane>>4.  Element e -> k = (e&7) + 8h + 16*(e>>3)  => two 16B loads at
// k-offsets {8h, 16+8h}.  ld must be a multiple of 8 elements.
__device__ __forceinline__ bf16x16 load_frag_a(const unsigned short* base,
                                               int row, int ld, int h) {
  FragU f;
  const unsigned short* p = base + row * ld + 8 * h;
  f.q[0] = *(const u32x4*)(p);
  f.q[1] = *(const u32x4*)(p + 16);
  return f.v;
}

// B-operand (32x16 bf16) loaded from an [N][K]-layout (transposed) tile.
// lane: n=lane&15, h=lane>>4.  Element e -> k = (e&7) + 8h + 16*(e>>3)  => two 16B loads.
__device__ __forceinline__ bf16x16 load_frag_b(const unsigned short* base,
                                               int row, int ld, int h) {
  FragU f;
  const unsigned short* p = base + row * ld + 8 * h;
  f.q[0] = *(const u32x4*)(p);
  f.q[1] = *(const u32x4*)(p + 16);
  return f.v;
}

// ---------------------------------------------------------------------------
// Kernel 1: f32 -> bf16 conversion of x (8 elements / thread)
// ---------------------------------------------------------------------------
__global__ void cvt_bf_kernel(const float* __restrict__ x,
                              unsigned short* __restrict__ xb, int n8) {
  int i = blockIdx.x * blockDim.x + threadIdx.x;
  if (i >= n8) return;
  const f32x4* s = (const f32x4*)x + (size_t)i * 2;
  f32x4 a = s[0], b = s[1];
  u32x4 r;
  r[0] = (unsigned)f2bf(a[0]) | ((unsigned)f2bf(a[1]) << 16);
  r[1] = (unsigned)f2bf(a[2]) | ((unsigned)f2bf(a[3]) << 16);
  r[2] = (unsigned)f2bf(b[0]) | ((unsigned)f2bf(b[1]) << 16);
  r[3] = (unsigned)f2bf(b[2]) | ((unsigned)f2bf(b[3]) << 16);
  ((u32x4*)xb)[i] = r;
}

// ---------------------------------------------------------------------------
// Kernel 2: weight generation (tiny: 4 x 512x512x128).
//   wqT[i][j] = dot(q_coeffs[i,:], basis[j,:])   (so x@q_w has B-op wqT[n][k])
//   same for k,v;  woT[i][j] = dot(basis[i,:], o_coeffs[j,:])
// ---------------------------------------------------------------------------
__global__ void gen_w_kernel(const float* __restrict__ basis,
                             const float* __restrict__ qc,
                             const float* __restrict__ kc,
                             const float* __restrict__ vc,
                             const float* __restrict__ oc,
                             unsigned short* __restrict__ wq,
                             unsigned short* __restrict__ wk,
                             unsigned short* __restrict__ wv,
                             unsigned short* __restrict__ wo) {
  int mat = blockIdx.y;
  int idx = blockIdx.x * 256 + threadIdx.x;
  int i = idx >> 9, j = idx & 511;
  const float *P, *Q;
  unsigned short* O;
  if (mat == 0)      { P = qc;    Q = basis; O = wq; }
  else if (mat == 1) { P = kc;    Q = basis; O = wk; }
  else if (mat == 2) { P = vc;    Q = basis; O = wv; }
  else               { P = basis; Q = oc;    O = wo; }
  const f32x4* p4 = (const f32x4*)(P + (size_t)i * 128);
  const f32x4* q4 = (const f32x4*)(Q + (size_t)j * 128);
  float s = 0.f;
#pragma unroll 8
  for (int n = 0; n < 32; n++) {
    f32x4 a = p4[n], b = q4[n];
    s += a[0] * b[0] + a[1] * b[1] + a[2] * b[2] + a[3] * b[3];
  }
  O[(size_t)i * 512 + j] = f2bf(s);
}

// ---------------------------------------------------------------------------
// Kernel 3: QKV projections.  C[m][n] = sum_k x[m][k] * W[k][n], W from Wt[n][k].
// Block tile 128x128, 8 waves as 4(m) x 2(n), K-loop of 32, double-buffered
// LDS fed by async DMA.  Q/K stored row-major [B*T][C]; V transposed [B][C][T].
// ---------------------------------------------------------------------------
__global__ __launch_bounds__(256) void gemm_qkv_kernel(
    const unsigned short* __restrict__ xb,
    const unsigned short* __restrict__ wq,
    const unsigned short* __restrict__ wk,
    const unsigned short* __restrict__ wv,
    unsigned short* __restrict__ qb,
    unsigned short* __restrict__ kb,
    unsigned short* __restrict__ vt) {
  __shared__ __align__(16) unsigned short As[2][128 * 48];
  __shared__ __align__(16) unsigned short Bs[2][128 * 48];

  const int m0 = blockIdx.x * 128;
  const int wsel = blockIdx.y >> 2;
  const int n0 = (blockIdx.y & 3) * 128;
  const unsigned short* wt = (wsel == 0) ? wq : ((wsel == 1) ? wk : wv);

  const int tid = threadIdx.x;
  const int lane = tid & 31, wid = tid >> 5;
  const int wr = wid >> 1, wc = wid & 1;
  const int ln = lane & 15, h = lane >> 4;

  f32x8 acc[2][4] = {};

  const int lrow = tid >> 1, lk0 = (tid & 1) * 16;
  const unsigned short* asrc = xb + ((size_t)(m0 + lrow)) * CDIM + lk0;
  const unsigned short* bsrc = wt + ((size_t)(n0 + lrow)) * CDIM + lk0;
  const int ldst = lrow * 48 + lk0;

  // prologue: tile 0 -> buffer 0
  async_copy16(&As[0][ldst],     asrc);
  async_copy16(&As[0][ldst + 8], asrc + 8);
  async_copy16(&Bs[0][ldst],     bsrc);
  async_copy16(&Bs[0][ldst + 8], bsrc + 8);

  for (int kk = 0; kk < 16; kk++) {
    const int cur = kk & 1;
    if (kk + 1 < 16) {
      const int nb = cur ^ 1, ko = (kk + 1) * 32;
      async_copy16(&As[nb][ldst],     asrc + ko);
      async_copy16(&As[nb][ldst + 8], asrc + ko + 8);
      async_copy16(&Bs[nb][ldst],     bsrc + ko);
      async_copy16(&Bs[nb][ldst + 8], bsrc + ko + 8);
      WAIT_ASYNC("0x4");   // previous tile fully landed (in-order retire)
    } else {
      WAIT_ASYNC("0x0");
    }
    __syncthreads();
    bf16x16 a0 = load_frag_a(&As[cur][(32 * wr) * 48], ln, 48, h);
    bf16x16 a1 = load_frag_a(&As[cur][(32 * wr + 16) * 48], ln, 48, h);
#pragma unroll
    for (int j = 0; j < 4; j++) {
      bf16x16 bb = load_frag_b(&Bs[cur][(64 * wc + 16 * j) * 48], ln, 48, h);
      acc[0][j] = wmma_bf16(a0, bb, acc[0][j]);
      acc[1][j] = wmma_bf16(a1, bb, acc[1][j]);
    }
    __syncthreads();
  }

  if (wsel < 2) {
    unsigned short* out = (wsel == 0) ? qb : kb;
#pragma unroll
    for (int a = 0; a < 2; a++)
#pragma unroll
      for (int j = 0; j < 4; j++) {
        int m = m0 + 32 * wr + 16 * a + 8 * h;
        int n = n0 + 64 * wc + 16 * j + ln;
#pragma unroll
        for (int d = 0; d < 8; d++)
          out[(size_t)(m + d) * CDIM + n] = f2bf(acc[a][j][d]);
      }
  } else {
    const int bidx = m0 >> 12;              // batch (tiles never cross T=4096)
    const int tb = m0 & 4095;
#pragma unroll
    for (int a = 0; a < 2; a++)
#pragma unroll
      for (int j = 0; j < 4; j++) {
        int tl = tb + 32 * wr + 16 * a + 8 * h;   // 8 consecutive t per lane
        int c = n0 + 64 * wc + 16 * j + ln;
        u32x4 pk;
#pragma unroll
        for (int q = 0; q < 4; q++)
          pk[q] = (unsigned)f2bf(acc[a][j][2 * q]) |
                  ((unsigned)f2bf(acc[a][j][2 * q + 1]) << 16);
        *(u32x4*)(vt + ((size_t)(bidx * CDIM + c)) * T_SEQ + tl) = pk;
      }
  }
}

// ---------------------------------------------------------------------------
// Kernel 4: decay-weighted attention (flash style, anti-causal band).
// retrieved[t,:] = sum_{s>t} decay^(s-t-1) * (q_t . k_s) * v_s
// Block: 64 t-rows x C=512, 8 waves = 4(t) x 2(c-half of 256).
// K/V tiles double-buffered and fed by async DMA; the two c-half waves split
// score columns, weighted scores staged in LDS as bf16, then 16 PV WMMAs per
// wave.  Early break when decay^(s0-t_max-1) < 2^-100 (invisible in f32).
// ---------------------------------------------------------------------------
__global__ __launch_bounds__(256) void attn_kernel(
    const unsigned short* __restrict__ qb,
    const unsigned short* __restrict__ kb,
    const unsigned short* __restrict__ vt,
    unsigned short* __restrict__ rb,
    const float* __restrict__ decay_logit_p) {
  constexpr int QLD = 528, KLD = 528, VLD = 32, SLD = 48;
  __shared__ __align__(16) unsigned short Qs[64 * QLD];        // ~66 KB
  __shared__ __align__(16) unsigned short Ks[2][32 * KLD];     // ~67 KB
  __shared__ __align__(16) unsigned short Vs[2][CDIM * VLD];   //  64 KB
  __shared__ __align__(16) unsigned short Ss[4 * 16 * SLD];    //   6 KB

  const int t0 = blockIdx.x * 64;
  const int b = blockIdx.y;
  const int tid = threadIdx.x;
  const int lane = tid & 31, wid = tid >> 5;
  const int wr = wid >> 1, wc = wid & 1;
  const int ln = lane & 15, h = lane >> 4;

  const float dl = decay_logit_p[0];
  const float decay = 1.0f / (1.0f + __expf(-dl));
  const float log2d = __log2f(decay);

  // Q tile (64 x 512) via async DMA, issued before everything else
  {
    const int row = tid >> 2, c0 = (tid & 3) * 128;
    const unsigned short* src = qb + ((size_t)(b * T_SEQ + t0 + row)) * CDIM + c0;
    unsigned short* dst = &Qs[row * QLD + c0];
#pragma unroll
    for (int i = 0; i < 16; i++) async_copy16(dst + i * 8, src + i * 8);
  }

  f32x8 acc[16] = {};

  int s_end;
  {
    float margin = (log2d < -1e-9f) ? (-100.0f / log2d) : (float)T_SEQ;
    long e = (long)t0 + 96 + (long)margin;
    if (e > T_SEQ) e = T_SEQ;
    s_end = ((int)e + 31) & ~31;
    if (s_end > T_SEQ) s_end = T_SEQ;
  }

  const int krow = tid >> 3, kc0 = (tid & 7) * 64;
  const int vrow = tid * 2;
  auto issue_kv = [&](int s0, int bsel) {
    const unsigned short* ksrc =
        kb + ((size_t)(b * T_SEQ + s0 + krow)) * CDIM + kc0;
    unsigned short* kdst = &Ks[bsel][krow * KLD + kc0];
#pragma unroll
    for (int i = 0; i < 8; i++) async_copy16(kdst + i * 8, ksrc + i * 8);
#pragma unroll
    for (int r = 0; r < 2; r++) {
      const unsigned short* vsrc =
          vt + ((size_t)(b * CDIM + vrow + r)) * T_SEQ + s0;
      unsigned short* vdst = &Vs[bsel][(vrow + r) * VLD];
#pragma unroll
      for (int i = 0; i < 4; i++) async_copy16(vdst + i * 8, vsrc + i * 8);
    }
  };

  issue_kv(t0, 0);   // tile 0 -> buffer 0

  int ti = 0;
  for (int s0 = t0; s0 < s_end; s0 += 32, ti++) {
    const int cur = ti & 1;
    if (s0 + 32 < s_end) {
      issue_kv(s0 + 32, cur ^ 1);
      WAIT_ASYNC("0x10");   // Q + tile ti landed (16 newer in flight)
    } else {
      WAIT_ASYNC("0x0");
    }
    __syncthreads();

    // scores: wave (wr,wc) computes 16(t) x 16(s) columns [16wc,16wc+16)
    f32x8 sc = {};
#pragma unroll
    for (int kk = 0; kk < 16; kk++) {
      bf16x16 af = load_frag_a(&Qs[(16 * wr) * QLD + kk * 32], ln, QLD, h);
      bf16x16 bf = load_frag_b(&Ks[cur][(16 * wc) * KLD + kk * 32], ln, KLD, h);
      sc = wmma_bf16(af, bf, sc);
    }
    // decay weighting + stage as bf16 A-tile [16 t][32 s]
    {
      const int sg = s0 + 16 * wc + ln;
      const int tgb = t0 + 16 * wr + 8 * h;
#pragma unroll
      for (int d = 0; d < 8; d++) {
        int t = tgb + d;
        float w = (sg > t) ? exp2f(log2d * (float)(sg - t - 1)) : 0.0f;
        Ss[(wr * 16 + 8 * h + d) * SLD + 16 * wc + ln] = f2bf(sc[d] * w);
      }
    }
    __syncthreads();

    // PV: acc[j] += Sw(16x32) x V(32x16)
    bf16x16 afr = load_frag_a(&Ss[(wr * 16) * SLD], ln, SLD, h);
#pragma unroll
    for (int j = 0; j < 16; j++) {
      bf16x16 bfv = load_frag_b(&Vs[cur][(256 * wc + 16 * j) * VLD], ln, VLD, h);
      acc[j] = wmma_bf16(afr, bfv, acc[j]);
    }
    __syncthreads();
  }

  // write retrieved bf16, row-major [B*T][C]
#pragma unroll
  for (int j = 0; j < 16; j++) {
    int c = 256 * wc + 16 * j + ln;
    size_t base = ((size_t)(b * T_SEQ + t0 + 16 * wr + 8 * h)) * CDIM + c;
#pragma unroll
    for (int d = 0; d < 8; d++)
      rb[base + (size_t)d * CDIM] = f2bf(acc[j][d]);
  }
}

// ---------------------------------------------------------------------------
// Kernel 5: out = (retrieved @ o_w^T) * out_scale, f32 output.
// Same tiling as kernel 3; B-operand from woT[n][k].
// ---------------------------------------------------------------------------
__global__ __launch_bounds__(256) void gemm_out_kernel(
    const unsigned short* __restrict__ rbuf,
    const unsigned short* __restrict__ wo,
    float* __restrict__ out,
    const float* __restrict__ out_scale_p) {
  __shared__ __align__(16) unsigned short As[2][128 * 48];
  __shared__ __align__(16) unsigned short Bs[2][128 * 48];

  const int m0 = blockIdx.x * 128;
  const int n0 = blockIdx.y * 128;
  const int tid = threadIdx.x;
  const int lane = tid & 31, wid = tid >> 5;
  const int wr = wid >> 1, wc = wid & 1;
  const int ln = lane & 15, h = lane >> 4;
  const float scale = out_scale_p[0];

  f32x8 acc[2][4] = {};

  const int lrow = tid >> 1, lk0 = (tid & 1) * 16;
  const unsigned short* asrc = rbuf + ((size_t)(m0 + lrow)) * CDIM + lk0;
  const unsigned short* bsrc = wo + ((size_t)(n0 + lrow)) * CDIM + lk0;
  const int ldst = lrow * 48 + lk0;

  async_copy16(&As[0][ldst],     asrc);
  async_copy16(&As[0][ldst + 8], asrc + 8);
  async_copy16(&Bs[0][ldst],     bsrc);
  async_copy16(&Bs[0][ldst + 8], bsrc + 8);

  for (int kk = 0; kk < 16; kk++) {
    const int cur = kk & 1;
    if (kk + 1 < 16) {
      const int nb = cur ^ 1, ko = (kk + 1) * 32;
      async_copy16(&As[nb][ldst],     asrc + ko);
      async_copy16(&As[nb][ldst + 8], asrc + ko + 8);
      async_copy16(&Bs[nb][ldst],     bsrc + ko);
      async_copy16(&Bs[nb][ldst + 8], bsrc + ko + 8);
      WAIT_ASYNC("0x4");
    } else {
      WAIT_ASYNC("0x0");
    }
    __syncthreads();
    bf16x16 a0 = load_frag_a(&As[cur][(32 * wr) * 48], ln, 48, h);
    bf16x16 a1 = load_frag_a(&As[cur][(32 * wr + 16) * 48], ln, 48, h);
#pragma unroll
    for (int j = 0; j < 4; j++) {
      bf16x16 bb = load_frag_b(&Bs[cur][(64 * wc + 16 * j) * 48], ln, 48, h);
      acc[0][j] = wmma_bf16(a0, bb, acc[0][j]);
      acc[1][j] = wmma_bf16(a1, bb, acc[1][j]);
    }
    __syncthreads();
  }

#pragma unroll
  for (int a = 0; a < 2; a++)
#pragma unroll
    for (int j = 0; j < 4; j++) {
      int m = m0 + 32 * wr + 16 * a + 8 * h;
      int n = n0 + 64 * wc + 16 * j + ln;
#pragma unroll
      for (int d = 0; d < 8; d++)
        out[(size_t)(m + d) * CDIM + n] = acc[a][j][d] * scale;
    }
}

// ---------------------------------------------------------------------------
extern "C" void kernel_launch(void* const* d_in, const int* in_sizes, int n_in,
                              void* d_out, int out_size, void* d_ws, size_t ws_size,
                              hipStream_t stream) {
  const float* x      = (const float*)d_in[0];
  const float* basis  = (const float*)d_in[1];
  const float* qc     = (const float*)d_in[2];
  const float* kc     = (const float*)d_in[3];
  const float* vc     = (const float*)d_in[4];
  const float* oc     = (const float*)d_in[5];
  const float* dlogit = (const float*)d_in[6];
  const float* oscale = (const float*)d_in[7];

  char* ws = (char*)d_ws;
  const size_t WSZ  = (size_t)512 * 512 * 2;            // 512 KB per weight
  const size_t XSZ  = (size_t)BATCH * T_SEQ * CDIM * 2; // 16 MB per tensor
  unsigned short* wq  = (unsigned short*)(ws);
  unsigned short* wk  = (unsigned short*)(ws + WSZ);
  unsigned short* wv  = (unsigned short*)(ws + 2 * WSZ);
  unsigned short* wo  = (unsigned short*)(ws + 3 * WSZ);
  unsigned short* xbf = (unsigned short*)(ws + 4 * WSZ);
  unsigned short* qbf = (unsigned short*)(ws + 4 * WSZ + XSZ);
  unsigned short* kbf = (unsigned short*)(ws + 4 * WSZ + 2 * XSZ);
  unsigned short* vtb = (unsigned short*)(ws + 4 * WSZ + 3 * XSZ);
  unsigned short* rbf = (unsigned short*)(ws + 4 * WSZ + 4 * XSZ);

  // 1. x -> bf16  (8.39M elements, 8 per thread)
  cvt_bf_kernel<<<4096, 256, 0, stream>>>(x, xbf, (BATCH * T_SEQ * CDIM) / 8);
  // 2. generate transposed projection weights (bf16)
  gen_w_kernel<<<dim3(1024, 4), 256, 0, stream>>>(basis, qc, kc, vc, oc,
                                                  wq, wk, wv, wo);
  // 3. Q, K (row-major) and V (transposed) projections
  gemm_qkv_kernel<<<dim3(128, 12), 256, 0, stream>>>(xbf, wq, wk, wv,
                                                     qbf, kbf, vtb);
  // 4. banded decay attention
  attn_kernel<<<dim3(T_SEQ / 64, BATCH), 256, 0, stream>>>(qbf, kbf, vtb,
                                                           rbf, dlogit);
  // 5. output projection + scale (f32)
  gemm_out_kernel<<<dim3(128, 4), 256, 0, stream>>>(rbf, wo, (float*)d_out,
                                                    oscale);
}
